// ExpEncoder_42460046688738
// MI455X (gfx1250) — matrix operand order repaired and checked by reading, not compiled, not run on hardware
//
#include <hip/hip_runtime.h>
#include <hip/hip_bf16.h>

typedef __attribute__((ext_vector_type(16))) __bf16 v16bf;
typedef __attribute__((ext_vector_type(8)))  float  v8f;

#define Bb 16
#define Pp 32
#define Gg 512
#define Dd 512
#define Aa 400
#define Hh 8
#define Apad 416   // 13 * 32

// -------- Kernel 1: proj[row, a] = sum_d emb_gene[idx[row], d] * w0[d, a] + b0[a]
// One wave per 16x16 output tile, K = 512 in steps of 32. w0 column panel staged
// into LDS with GLOBAL_LOAD_ASYNC_TO_LDS_B128 (ASYNCcnt DMA path).
__global__ __launch_bounds__(256) void k_proj(const int* __restrict__ omc_idx,
                                              const float* __restrict__ emb_gene,
                                              const float* __restrict__ w0,
                                              const float* __restrict__ b0,
                                              float* __restrict__ proj) {
  __shared__ int sidx[128];
  __shared__ __align__(16) float s_b[512 * 16];   // w0 panel [K=512][16] f32 (32 KB)

  const int t    = threadIdx.x;
  const int wave = t >> 5;
  const int lane = t & 31;
  const int half = lane >> 4;
  const int ln   = lane & 15;

  const int row_base = blockIdx.x * 128;   // 64 blocks * 128 rows = 8192
  const int col0     = blockIdx.y * 16;    // 25 blocks * 16 = 400

  if (t < 128) sidx[t] = omc_idx[row_base + t];

  // Async-DMA stage of the B panel: 512 rows * 64 B = 2048 x b128, 8 per thread.
  {
    const float* gbase = w0 + col0;
#pragma unroll
    for (int j = 0; j < 8; ++j) {
      int cchunk = t + j * 256;                 // [0, 2048)
      int k  = cchunk >> 2;
      int n4 = (cchunk & 3) * 4;
      unsigned lds_addr = (unsigned)(uintptr_t)(&s_b[k * 16 + n4]);
      const float* gp = gbase + (size_t)k * Aa + n4;
      asm volatile("global_load_async_to_lds_b128 %0, %1, off"
                   :: "v"(lds_addr), "v"(gp) : "memory");
    }
    asm volatile("s_wait_asynccnt 0x0" ::: "memory");
  }
  __syncthreads();

  const int local_row = wave * 16 + ln;                  // A-matrix row for this lane
  const float* arow = emb_gene + (size_t)sidx[local_row] * Dd;

  v8f c = {};
  for (int kk = 0; kk < Dd; kk += 32) {
    v16bf a, bmat;
#pragma unroll
    for (int e = 0; e < 16; ++e) {
      // A in this code's contiguous k order (same as B): lanes 0-15 hold K 0..15; lanes 16-31 hold K 16..31
      int kl = half * 16 + e;
      a[e] = (__bf16)arow[kk + kl];
      // B in the same contiguous k order as A: lanes 0-15 -> K 0..15, lanes 16-31 -> K 16..31
      bmat[e] = (__bf16)s_b[(kk + half * 16 + e) * 16 + ln];
    }
    c = __builtin_amdgcn_wmma_f32_16x16x32_bf16(false, a, false, bmat, (short)0, c,
                                                false, false);
  }

  const float bias = b0[col0 + ln];
#pragma unroll
  for (int r = 0; r < 8; ++r) {
    int m   = r + half * 8;                 // C/D layout: VGPR r -> M=r (lanes 0-15) / M=8+r
    int row = row_base + wave * 16 + m;
    proj[(size_t)row * Aa + col0 + ln] = c[r] + bias;
  }
}

// -------- Kernel 2: fused tanh-attention, logits via WMMA, softmax over G, sum over H.
// One block per (b,p); 8 waves, each wave owns 4 M-tiles of 16 genes.
// A-frag (16x32 bf16) = tanh(proj + ctx) built directly in fragment layout;
// B-frag = beta_w padded [416 x 16] (zero rows k>=400 kill garbage A there).
// beta_b is a per-head constant -> cancels in the softmax over genes.
__global__ __launch_bounds__(256) void k_attn(const float* __restrict__ proj,
                                              const int* __restrict__ ptw_ids,
                                              const float* __restrict__ emb_ptw,
                                              const float* __restrict__ beta_w,
                                              float* __restrict__ wout) {
  __shared__ __align__(16) float  s_ctx[Apad];
  __shared__ __bf16 s_bw[Apad * 16];     // [k][n] bf16, zero-padded (13.3 KB)
  __shared__ float  s_logits[Gg * Hh];   // [g][h] (16 KB)
  __shared__ float  s_red[256 * Hh];     // (8 KB)

  const int t    = threadIdx.x;
  const int wave = t >> 5;
  const int lane = t & 31;
  const int half = lane >> 4;
  const int ln   = lane & 15;

  const int bp = blockIdx.x;          // b*32 + p
  const int p  = bp & 31;
  const int b  = bp >> 5;

  const int pid = ptw_ids[p];
  for (int i = t; i < Apad; i += 256)
    s_ctx[i] = (i < Aa) ? emb_ptw[(size_t)pid * Aa + i] : 0.f;
  for (int i = t; i < Apad * 16; i += 256) {
    int k = i >> 4, n = i & 15;
    s_bw[i] = (k < Aa && n < Hh) ? (__bf16)beta_w[k * Hh + n] : (__bf16)0.f;
  }
  __syncthreads();

  for (int mt = wave; mt < 32; mt += 8) {
    const int g0 = mt * 16;
    const float* prow = proj + (size_t)(b * Gg + g0 + ln) * Aa;
    v8f c = {};
    int kk = 0;
    for (; kk < 384; kk += 32) {
      const float4* p4 = (const float4*)(prow + kk + half * 8);
      const float4* q4 = (const float4*)(prow + kk + 16 + half * 8);
      const float4* c4 = (const float4*)(s_ctx + kk + half * 8);
      const float4* d4 = (const float4*)(s_ctx + kk + 16 + half * 8);
      float4 pa = p4[0], pb = p4[1], qa = q4[0], qb = q4[1];
      float4 ca = c4[0], cb = c4[1], da = d4[0], db = d4[1];
      float xs[16] = {pa.x + ca.x, pa.y + ca.y, pa.z + ca.z, pa.w + ca.w,
                      pb.x + cb.x, pb.y + cb.y, pb.z + cb.z, pb.w + cb.w,
                      qa.x + da.x, qa.y + da.y, qa.z + da.z, qa.w + da.w,
                      qb.x + db.x, qb.y + db.y, qb.z + db.z, qb.w + db.w};
      v16bf a, bm;
#pragma unroll
      for (int e = 0; e < 16; ++e) {
        // tanh(x) = 1 - 2/(exp2(2*log2e*x)+1)  -> v_exp_f32 + v_rcp_f32
        float e2 = __builtin_amdgcn_exp2f(xs[e] * 2.88539008f);
        a[e]  = (__bf16)(1.f - 2.f * __builtin_amdgcn_rcpf(e2 + 1.f));
        bm[e] = s_bw[(kk + (e & 7) + half * 8 + ((e & 8) << 1)) * 16 + ln];
      }
      c = __builtin_amdgcn_wmma_f32_16x16x32_bf16(false, a, false, bm, (short)0, c,
                                                  false, false);
    }
    { // tail kk=384: run1 (k = 384+half*8+e, e<8) fully valid; run2 (>=400) zeroed
      const float4* p4 = (const float4*)(prow + kk + half * 8);
      const float4* c4 = (const float4*)(s_ctx + kk + half * 8);
      float4 pa = p4[0], pb = p4[1], ca = c4[0], cb = c4[1];
      float xs[8] = {pa.x + ca.x, pa.y + ca.y, pa.z + ca.z, pa.w + ca.w,
                     pb.x + cb.x, pb.y + cb.y, pb.z + cb.z, pb.w + cb.w};
      v16bf a, bm;
#pragma unroll
      for (int e = 0; e < 8; ++e) {
        float e2 = __builtin_amdgcn_exp2f(xs[e] * 2.88539008f);
        a[e]     = (__bf16)(1.f - 2.f * __builtin_amdgcn_rcpf(e2 + 1.f));
        a[e + 8] = (__bf16)0.f;
      }
#pragma unroll
      for (int e = 0; e < 16; ++e) bm[e] = s_bw[(kk + (e & 7) + half * 8 + ((e & 8) << 1)) * 16 + ln];
      c = __builtin_amdgcn_wmma_f32_16x16x32_bf16(false, a, false, bm, (short)0, c,
                                                  false, false);
    }
    if (ln < Hh) {
#pragma unroll
      for (int r = 0; r < 8; ++r) {
        int g = g0 + r + half * 8;
        s_logits[g * Hh + ln] = c[r];
      }
    }
  }
  __syncthreads();

  // Softmax over 512 genes per head, then sum over heads.
  float l0[Hh], l1[Hh];
#pragma unroll
  for (int h = 0; h < Hh; ++h) {
    l0[h] = s_logits[t * Hh + h];
    l1[h] = s_logits[(t + 256) * Hh + h];
  }
#pragma unroll
  for (int h = 0; h < Hh; ++h) s_red[t * Hh + h] = fmaxf(l0[h], l1[h]);
  __syncthreads();
  for (int s = 128; s > 0; s >>= 1) {
    if (t < s) {
#pragma unroll
      for (int h = 0; h < Hh; ++h)
        s_red[t * Hh + h] = fmaxf(s_red[t * Hh + h], s_red[(t + s) * Hh + h]);
    }
    __syncthreads();
  }
  float mh[Hh];
#pragma unroll
  for (int h = 0; h < Hh; ++h) mh[h] = s_red[h];
  __syncthreads();

  float e0[Hh], e1[Hh];
#pragma unroll
  for (int h = 0; h < Hh; ++h) {
    e0[h] = __builtin_amdgcn_exp2f((l0[h] - mh[h]) * 1.44269504f);
    e1[h] = __builtin_amdgcn_exp2f((l1[h] - mh[h]) * 1.44269504f);
    s_red[t * Hh + h] = e0[h] + e1[h];
  }
  __syncthreads();
  for (int s = 128; s > 0; s >>= 1) {
    if (t < s) {
#pragma unroll
      for (int h = 0; h < Hh; ++h) s_red[t * Hh + h] += s_red[(t + s) * Hh + h];
    }
    __syncthreads();
  }
  float wa = 0.f, wb = 0.f;
#pragma unroll
  for (int h = 0; h < Hh; ++h) {
    float inv = __builtin_amdgcn_rcpf(s_red[h]);
    wa += e0[h] * inv;
    wb += e1[h] * inv;
  }
  wout[(size_t)bp * Gg + t]       = wa;
  wout[(size_t)bp * Gg + t + 256] = wb;
}

// -------- Kernel 3: out[b,p,d] = sum_g w[b,p,g] * emb_gene[idx[b,g], d]
// Per b: [32,512] x [512,512]. One wave per 16x16 tile of [P, D].
__global__ __launch_bounds__(256) void k_out(const float* __restrict__ wbuf,
                                             const int* __restrict__ omc_idx,
                                             const float* __restrict__ emb_gene,
                                             float* __restrict__ out) {
  __shared__ int sidx[Gg];
  const int t    = threadIdx.x;
  const int wave = t >> 5;
  const int lane = t & 31;
  const int half = lane >> 4;
  const int ln   = lane & 15;

  const int bm = blockIdx.x;         // b*2 + mtile  (32 blocks)
  const int b  = bm >> 1;
  const int p0 = (bm & 1) * 16;

  for (int i = t; i < Gg; i += 256) sidx[i] = omc_idx[b * Gg + i];
  __syncthreads();

  const int d0 = (blockIdx.y * 8 + wave) * 16;       // 4*8 = 32 n-tiles -> D = 512
  const float* wrow = wbuf + (size_t)(b * Pp + p0 + ln) * Gg;

  v8f c = {};
  for (int kk = 0; kk < Gg; kk += 32) {
    v16bf a, bmat;
#pragma unroll
    for (int e = 0; e < 16; ++e) {
      int kl = half * 16 + e;
      a[e] = (__bf16)wrow[kk + kl];
      bmat[e] = (__bf16)emb_gene[(size_t)sidx[kk + half * 16 + e] * Dd + d0 + ln];
    }
    c = __builtin_amdgcn_wmma_f32_16x16x32_bf16(false, a, false, bmat, (short)0, c,
                                                false, false);
  }
#pragma unroll
  for (int r = 0; r < 8; ++r) {
    int m = r + half * 8;
    out[(size_t)(b * Pp + p0 + m) * Dd + d0 + ln] = c[r];
  }
}

extern "C" void kernel_launch(void* const* d_in, const int* in_sizes, int n_in,
                              void* d_out, int out_size, void* d_ws, size_t ws_size,
                              hipStream_t stream) {
  const int*   omc_idx  = (const int*)d_in[0];     // [B, G]
  const int*   ptw_ids  = (const int*)d_in[1];     // [1, P]
  const float* emb_gene = (const float*)d_in[2];   // [OMC+1, D]
  const float* emb_ptw  = (const float*)d_in[3];   // [PTW, A]
  const float* w0       = (const float*)d_in[4];   // [D, A]
  const float* b0       = (const float*)d_in[5];   // [A]
  const float* beta_w   = (const float*)d_in[6];   // [A, H]
  // d_in[7] = beta_b: per-head constant, cancels in softmax over genes.

  float* proj = (float*)d_ws;                                        // [B*G, A] f32 (13.1 MB)
  float* wbuf = (float*)((char*)d_ws + (size_t)Bb * Gg * Aa * 4);    // [B*P, G] f32 (1 MB)
  float* outp = (float*)d_out;                                       // [B, P, D] f32

  k_proj<<<dim3(64, 25), 256, 0, stream>>>(omc_idx, emb_gene, w0, b0, proj);
  k_attn<<<dim3(512), 256, 0, stream>>>(proj, ptw_ids, emb_ptw, beta_w, wbuf);
  k_out<<<dim3(32, 4), 256, 0, stream>>>(wbuf, omc_idx, emb_gene, outp);
}
